// EEGEncoder_rwkv_60129542391
// MI455X (gfx1250) — matrix operand order repaired and checked by reading, not compiled, not run on hardware
//
#include <hip/hip_runtime.h>

typedef float v8f  __attribute__((ext_vector_type(8)));
typedef __bf16 v16bf __attribute__((ext_vector_type(16)));

#define B_   8
#define E_   128
#define T0_  29184
#define FC_  64
#define EC_  128
#define TP_  7295
#define TPs_ 7296
#define NE_  64
#define HID_ 256

static __device__ __forceinline__ __bf16 f2b(float x) { return (__bf16)x; }
static __device__ __forceinline__ float  b2f(__bf16 x) { return (float)x; }

// ---------------- Stage 1: BatchNorm over time-index "channels" ----------------
__global__ void k_tstats(const float* __restrict__ spike, float* mean, float* rstd) {
    int t = blockIdx.x * blockDim.x + threadIdx.x;
    if (t >= T0_) return;
    float s = 0.f, s2 = 0.f;
    for (int r = 0; r < B_ * E_; ++r) {
        float v = spike[(size_t)r * T0_ + t];
        s += v; s2 += v * v;
    }
    float m = s * (1.f / (B_ * E_));
    float var = s2 * (1.f / (B_ * E_)) - m * m;
    mean[t] = m;
    rstd[t] = rsqrtf(var + 1e-5f);
}

__global__ void k_bn1_apply(const float* __restrict__ spike, const float* mean, const float* rstd,
                            const float* g, const float* b, __bf16* xbf) {
    size_t i = (size_t)blockIdx.x * blockDim.x + threadIdx.x;
    if (i >= (size_t)B_ * E_ * T0_) return;
    int t = (int)(i % T0_);
    float v = (spike[i] - mean[t]) * rstd[t] * g[t] + b[t];
    xbf[i] = f2b(v);
}

// ---------------- Chebyshev prep: M = gcw0 + gcw1@L + gcw2@(2LL - I) ----------------
__global__ void k_cheb(const float* __restrict__ A, const float* __restrict__ gcw,
                       const float* __restrict__ gcb,
                       float* Lm, float* T2m, float* Mm, float* gbias) {
    __shared__ float dinv[E_];
    int tid = threadIdx.x;
    if (tid < E_) {
        float s = 0.f;
        for (int j = 0; j < E_; ++j) s += fmaxf(A[tid * E_ + j], 0.f);
        dinv[tid] = rsqrtf(s + 1e-10f);
    }
    __syncthreads();
    for (int idx = tid; idx < E_ * E_; idx += blockDim.x) {
        int i = idx >> 7, j = idx & 127;
        Lm[idx] = dinv[i] * fmaxf(A[idx], 0.f) * dinv[j];
    }
    __threadfence(); __syncthreads();
    for (int idx = tid; idx < E_ * E_; idx += blockDim.x) {
        int i = idx >> 7, j = idx & 127;
        float s = 0.f;
        for (int u = 0; u < E_; ++u) s += Lm[i * E_ + u] * Lm[u * E_ + j];
        T2m[idx] = 2.f * s - (i == j ? 1.f : 0.f);
    }
    __threadfence(); __syncthreads();
    for (int idx = tid; idx < E_ * E_; idx += blockDim.x) {
        int i = idx >> 7, j = idx & 127;
        float s = gcw[idx];
        const float* w1 = gcw + E_ * E_;
        const float* w2 = gcw + 2 * E_ * E_;
        for (int u = 0; u < E_; ++u)
            s += w1[i * E_ + u] * Lm[u * E_ + j] + w2[i * E_ + u] * T2m[u * E_ + j];
        Mm[idx] = s;
    }
    if (tid < E_) gbias[tid] = gcb[tid] + gcb[E_ + tid] + gcb[2 * E_ + tid];
}

// ---------------- WMMA GEMM: Y[b,m,t] = act(sum_k W[m,k] X[b,k,t] + bias) (+addin) ----------------
// 4 waves / block; block tile = 16 rows x 64 cols; padded stride Ts (mult of 64) -> no edge path,
// B staging = 2x 16B vector loads per thread; fragments read as 32B LDS vectors.
template <int KT, bool WT>
__global__ void k_wmma_gemm(const float* __restrict__ Wf,
                            const __bf16* __restrict__ X,
                            const float* __restrict__ bias,
                            const float* __restrict__ addin,
                            float* Yf, __bf16* Ybf,
                            int M, int T, int Ts, int act) {
    __shared__ alignas(32) __bf16 AfS[32 * 16];      // fragment order: [lane][e]
    __shared__ alignas(32) __bf16 BfS[4][32 * 16];   // per-wave fragment order
    const int tid  = threadIdx.x;
    const int lane = tid & 31;
    const int wv   = tid >> 5;
    const int t0 = blockIdx.x * 64;
    const int m0 = blockIdx.y * 16;
    const int b  = blockIdx.z;
    const size_t xbase = (size_t)b * KT * Ts;
    // per-thread staging roles
    const int am = tid >> 3;          // A: row 0..15
    const int ak = (tid & 7) * 4;     // A: 4 consecutive k
    const int bk = tid >> 2;          // B: k row 0..31
    const int bg = tid & 3;           // B: wave-column group
    const int laneB = 16 * ((bk >> 3) & 1);
    const int be = (bk & 7) | ((bk >> 4) << 3);
    v8f acc = {};
#pragma unroll
    for (int k0 = 0; k0 < KT; k0 += 32) {
        // --- stage A (16 x 32) into fragment order ---
#pragma unroll
        for (int q = 0; q < 4; ++q) {
            int k = ak + q;
            float w = WT ? Wf[(size_t)(k0 + k) * M + (m0 + am)]
                         : Wf[(size_t)(m0 + am) * KT + (k0 + k)];
            int h = (k >> 3) & 1;
            AfS[(am + 16 * h) * 16 + ((k & 7) | ((k >> 4) << 3))] = f2b(w);
        }
        // --- stage B (32 k x 64 cols): two aligned 16B loads, scatter to fragment order ---
        {
            const __bf16* src = X + xbase + (size_t)(k0 + bk) * Ts + t0 + bg * 16;
            if (k0 + 32 < KT)
                __builtin_prefetch((const void*)(src + (size_t)32 * Ts), 0, 1);
            union { uint4 u[2]; __bf16 h[16]; } bufB;
            const uint4* p = (const uint4*)src;
            bufB.u[0] = p[0];
            bufB.u[1] = p[1];
#pragma unroll
            for (int nn = 0; nn < 16; ++nn)
                BfS[bg][(laneB + nn) * 16 + be] = bufB.h[nn];
        }
        __syncthreads();
        v16bf a  = ((const v16bf*)AfS)[lane];
        v16bf bb = ((const v16bf*)BfS[wv])[lane];
        acc = __builtin_amdgcn_wmma_f32_16x16x32_bf16(false, a, false, bb, (short)0, acc, false, false);
        __syncthreads();
    }
    const int half = lane >> 4, l16 = lane & 15;
#pragma unroll
    for (int i = 0; i < 8; ++i) {
        int m = m0 + i + 8 * half;
        int t = t0 + wv * 16 + l16;
        if (t < T) {
            float v = acc[i];
            if (bias) v += bias[m];
            if (act == 1) v = fmaxf(v, 0.f);
            else if (act == 2) v = 1.f / (1.f + __expf(-v));
            else if (act == 3) { v = fmaxf(v, 0.f); v = v * v; }
            size_t o = (size_t)b * M * Ts + (size_t)m * Ts + t;
            if (addin) v += addin[o];
            if (Yf)  Yf[o]  = v;
            if (Ybf) Ybf[o] = f2b(v);
        }
    }
}

// ---------------- Projection conv (128 -> 64, k=8, s=4) as im2col WMMA GEMM ----------------
// B staging uses contiguous-span vector loads: for a k-slab, only 4 channel rows are touched and
// the strided gather 4t+j lives inside one contiguous span per row -> 2x uint4 loads/thread.
__global__ void k_wmma_proj(const float* __restrict__ Wf,     // [64][1024]
                            const __bf16* __restrict__ G,     // [b][128][T0]
                            float* Yf) {                      // [b][64][TPs]
    __shared__ alignas(32) __bf16 AfS[32 * 16];
    __shared__ alignas(32) __bf16 BfS[4][32 * 16];
    const int tid  = threadIdx.x;
    const int lane = tid & 31;
    const int wv   = tid >> 5;
    const int t0 = blockIdx.x * 64;
    const int m0 = blockIdx.y * 16;
    const int b  = blockIdx.z;
    const size_t gbase = (size_t)b * E_ * T0_;
    const int am = tid >> 3;
    const int ak = (tid & 7) * 4;
    const int cp = tid >> 5;          // channel-in-slab 0..3
    const int sp = tid & 31;          // span index
    const int bgp = (2 * sp) >> 4;    // column group for this thread's two t's
    v8f acc = {};
#pragma unroll 4
    for (int k0 = 0; k0 < 1024; k0 += 32) {
#pragma unroll
        for (int q = 0; q < 4; ++q) {
            int k = ak + q;
            float w = Wf[(size_t)(m0 + am) * 1024 + (k0 + k)];
            int h = (k >> 3) & 1;
            AfS[(am + 16 * h) * 16 + ((k & 7) | ((k >> 4) << 3))] = f2b(w);
        }
        {
            const int vch = (k0 >> 3) + cp;
            const __bf16* src = G + gbase + (size_t)vch * T0_ + 4 * t0 + sp * 8;
            if (k0 + 32 < 1024)
                __builtin_prefetch((const void*)(src + (size_t)4 * T0_), 0, 1);
            union { uint4 u[2]; __bf16 h[16]; } bufB;
            const uint4* p = (const uint4*)src;
            bufB.u[0] = p[0];
            bufB.u[1] = p[1];
#pragma unroll
            for (int dt = 0; dt < 2; ++dt) {
                int nn = (2 * sp + dt) & 15;
#pragma unroll
                for (int j = 0; j < 8; ++j) {
                    int kk = cp * 8 + j;
                    BfS[bgp][(16 * ((kk >> 3) & 1) + nn) * 16 + ((kk & 7) | ((kk >> 4) << 3))] = bufB.h[4 * dt + j];
                }
            }
        }
        __syncthreads();
        v16bf a  = ((const v16bf*)AfS)[lane];
        v16bf bb = ((const v16bf*)BfS[wv])[lane];
        acc = __builtin_amdgcn_wmma_f32_16x16x32_bf16(false, a, false, bb, (short)0, acc, false, false);
        __syncthreads();
    }
    const int half = lane >> 4, l16 = lane & 15;
#pragma unroll
    for (int i = 0; i < 8; ++i) {
        int m = m0 + i + 8 * half;
        int t = t0 + wv * 16 + l16;
        if (t < TP_)
            Yf[(size_t)b * 64 * TPs_ + (size_t)m * TPs_ + t] = acc[i];
    }
}

// ---------------- LayerNorm over channel dim (also used as cLN) ----------------
__global__ void k_lnch(const float* __restrict__ X, const float* g, const float* bb,
                       float* Yf, __bf16* Ybf, int C, int T, int Ts, int Bn) {
    int idx = blockIdx.x * blockDim.x + threadIdx.x;
    if (idx >= Bn * T) return;
    int b = idx / T, t = idx - b * T;
    const float* base = X + (size_t)b * C * Ts + t;
    float s = 0.f, s2 = 0.f;
    for (int c = 0; c < C; ++c) { float v = base[(size_t)c * Ts]; s += v; s2 += v * v; }
    float m = s / C;
    float r = rsqrtf(s2 / C - m * m + 1e-5f);
    for (int c = 0; c < C; ++c) {
        float v = (base[(size_t)c * Ts] - m) * r * g[c] + bb[c];
        size_t o = (size_t)b * C * Ts + (size_t)c * Ts + t;
        if (Yf)  Yf[o]  = v;
        if (Ybf) Ybf[o] = f2b(v);
    }
}

// ---------------- train-mode BN stats per channel over (B,T) ----------------
__global__ void k_chanstats(const float* __restrict__ X, float* mean, float* rstd,
                            int C, int T, int Ts, int Bn) {
    int c = blockIdx.x;
    int n = Bn * T;
    float s = 0.f, s2 = 0.f;
    for (int i = threadIdx.x; i < n; i += blockDim.x) {
        int b = i / T, t = i - b * T;
        float v = X[((size_t)b * C + c) * Ts + t];
        s += v; s2 += v * v;
    }
    __shared__ float sh[256], sh2[256];
    sh[threadIdx.x] = s; sh2[threadIdx.x] = s2;
    __syncthreads();
    for (int off = 128; off > 0; off >>= 1) {
        if ((int)threadIdx.x < off) { sh[threadIdx.x] += sh[threadIdx.x + off]; sh2[threadIdx.x] += sh2[threadIdx.x + off]; }
        __syncthreads();
    }
    if (threadIdx.x == 0) {
        float m = sh[0] / n;
        mean[c] = m;
        rstd[c] = rsqrtf(sh2[0] / n - m * m + 1e-5f);
    }
}

__global__ void k_bnprelu(const float* __restrict__ X, const float* mean, const float* rstd,
                          const float* g, const float* bb, const float* alpha,
                          __bf16* Z, int C, int T, int Ts, size_t N) {
    size_t i = (size_t)blockIdx.x * blockDim.x + threadIdx.x;
    if (i >= N) return;
    int t = (int)(i % T);
    size_t bc = i / T;
    int c = (int)(bc % C);
    size_t o = bc * Ts + t;
    float v = (X[o] - mean[c]) * rstd[c] * g[c] + bb[c];
    float a = alpha[0];
    Z[o] = f2b(v >= 0.f ? v : a * v);
}

__global__ void k_resmerge(const float* __restrict__ T2x, const float* mean, const float* rstd,
                           const float* g, const float* bb, const float* __restrict__ dres,
                           const float* alpha, float* Y, __bf16* Ybf,
                           int C, int Tins, int To, int Tos, int Bn) {
    size_t i = (size_t)blockIdx.x * blockDim.x + threadIdx.x;
    size_t N = (size_t)Bn * C * To;
    if (i >= N) return;
    int to = (int)(i % To);
    size_t bc = i / To;
    int c = (int)(bc % C);
    size_t inb = bc * Tins + (size_t)3 * to;
    size_t ob  = bc * Tos + to;
    float a = alpha[0];
    float mx = -3.4e38f;
    for (int j = 0; j < 3; ++j) {
        float v = (T2x[inb + j] - mean[c]) * rstd[c] * g[c] + bb[c] + dres[inb + j];
        v = v >= 0.f ? v : a * v;
        mx = fmaxf(mx, v);
    }
    Y[ob] = mx;
    Ybf[ob] = f2b(mx);
}

// ---------------- Bi-WKV: one block per (b, channel) ----------------
__global__ void k_wkv(const float* __restrict__ kk, const float* __restrict__ vv,
                      const float* __restrict__ sr, const float* __restrict__ decay,
                      const float* __restrict__ first, __bf16* out, int C, int T, int Ts) {
    int b = blockIdx.x / C, c = blockIdx.x % C;
    size_t base = ((size_t)b * C + c) * Ts;
    float w = __expf(decay[c]) / (float)T;
    float u = first[c] / (float)T;
    float corr = __expf(u) - __expf(w);
    __shared__ float sek[272], sekv[272];
    for (int i = threadIdx.x; i < T; i += blockDim.x) {
        float ek = __expf(kk[base + i]);
        sek[i] = ek;
        sekv[i] = ek * vv[base + i];
    }
    __syncthreads();
    for (int t = threadIdx.x; t < T; t += blockDim.x) {
        float num = 0.f, den = 0.f;
        for (int i = 0; i < T; ++i) {
            float dist = -(fabsf((float)(t - i)) - 1.f);
            float D = __expf(dist * w);
            num += D * sekv[i];
            den += D * sek[i];
        }
        num += corr * sekv[t];
        den += corr * sek[t];
        float wkv = num / den;
        out[base + t] = f2b(sr[base + t] * wkv);
    }
}

__global__ void k_final(const float* h1, const float* rr2, const float* ffnv, float* out,
                        int T, int Ts, int N) {
    int i = blockIdx.x * blockDim.x + threadIdx.x;
    if (i >= N) return;
    int t = i % T;
    int bc = i / T;
    size_t o = (size_t)bc * Ts + t;
    out[i] = h1[o] + rr2[o] * ffnv[o];
}

// ==================================================================================
extern "C" void kernel_launch(void* const* d_in, const int* in_sizes, int n_in,
                              void* d_out, int out_size, void* d_ws, size_t ws_size,
                              hipStream_t stream) {
    const float* spike = (const float*)d_in[0];
    const float* bn1g = (const float*)d_in[1];
    const float* bn1b = (const float*)d_in[2];
    const float* Amat = (const float*)d_in[3];
    const float* gcw  = (const float*)d_in[4];
    const float* gcb  = (const float*)d_in[5];
    const float* projw = (const float*)d_in[6];
    const float* clng = (const float*)d_in[7];
    const float* clnb = (const float*)d_in[8];
    const float* inw  = (const float*)d_in[9];
    const float* inb  = (const float*)d_in[10];
    const float* r0c1 = (const float*)d_in[11]; const float* r0b1g = (const float*)d_in[12];
    const float* r0b1b = (const float*)d_in[13]; const float* r0c2 = (const float*)d_in[14];
    const float* r0b2g = (const float*)d_in[15]; const float* r0b2b = (const float*)d_in[16];
    const float* r0a1 = (const float*)d_in[17]; const float* r0a2 = (const float*)d_in[18];
    const float* r1c1 = (const float*)d_in[19]; const float* r1b1g = (const float*)d_in[20];
    const float* r1b1b = (const float*)d_in[21]; const float* r1c2 = (const float*)d_in[22];
    const float* r1b2g = (const float*)d_in[23]; const float* r1b2b = (const float*)d_in[24];
    const float* r1a1 = (const float*)d_in[25]; const float* r1a2 = (const float*)d_in[26];
    const float* r1dn = (const float*)d_in[27];
    const float* r2c1 = (const float*)d_in[28]; const float* r2b1g = (const float*)d_in[29];
    const float* r2b1b = (const float*)d_in[30]; const float* r2c2 = (const float*)d_in[31];
    const float* r2b2g = (const float*)d_in[32]; const float* r2b2b = (const float*)d_in[33];
    const float* r2a1 = (const float*)d_in[34]; const float* r2a2 = (const float*)d_in[35];
    const float* outw = (const float*)d_in[36]; const float* outb = (const float*)d_in[37];
    const float* ln0g = (const float*)d_in[38]; const float* ln0b = (const float*)d_in[39];
    const float* ln1g = (const float*)d_in[40]; const float* ln1b = (const float*)d_in[41];
    const float* ln2g = (const float*)d_in[42]; const float* ln2b = (const float*)d_in[43];
    const float* Wk = (const float*)d_in[44]; const float* Wv = (const float*)d_in[45];
    const float* Wr = (const float*)d_in[46]; const float* Wo = (const float*)d_in[47];
    const float* decay = (const float*)d_in[48]; const float* first = (const float*)d_in[49];
    const float* fWk = (const float*)d_in[50]; const float* fWv = (const float*)d_in[51];
    const float* fWr = (const float*)d_in[52];
    (void)in_sizes; (void)n_in; (void)out_size; (void)ws_size;

    // padded strides (multiples of 64)
    const int T1 = 2431, T1s = 2432;
    const int T2 = 810,  T2s = 832;
    const int TR = 270,  TRs = 320;
    const int NRlog = B_ * NE_ * TR;

    char* w = (char*)d_ws;
    auto alloc = [&](size_t bytes) { void* p = (void*)w; w += (bytes + 255) & ~(size_t)255; return p; };
    float*  mean_t = (float*)alloc((size_t)T0_ * 4);
    float*  rstd_t = (float*)alloc((size_t)T0_ * 4);
    __bf16* xbf    = (__bf16*)alloc((size_t)B_ * E_ * T0_ * 2);
    float*  Lm     = (float*)alloc(E_ * E_ * 4);
    float*  T2m    = (float*)alloc(E_ * E_ * 4);
    float*  Mm     = (float*)alloc(E_ * E_ * 4);
    float*  gbias  = (float*)alloc(E_ * 4);
    __bf16* goutbf = (__bf16*)alloc((size_t)B_ * E_ * T0_ * 2);
    float*  y1     = (float*)alloc((size_t)B_ * FC_ * TPs_ * 4);
    __bf16* ybf    = (__bf16*)alloc((size_t)B_ * FC_ * TPs_ * 2);
    float*  yA     = (float*)alloc((size_t)B_ * FC_ * TPs_ * 4);
    __bf16* yAbf   = (__bf16*)alloc((size_t)B_ * FC_ * TPs_ * 2);
    float*  yB     = (float*)alloc((size_t)B_ * FC_ * TPs_ * 4);
    __bf16* yBbf   = (__bf16*)alloc((size_t)B_ * FC_ * TPs_ * 2);
    float*  t1     = (float*)alloc((size_t)B_ * FC_ * TPs_ * 4);
    __bf16* zbf    = (__bf16*)alloc((size_t)B_ * FC_ * TPs_ * 2);
    float*  t2     = (float*)alloc((size_t)B_ * FC_ * TPs_ * 4);
    float*  dres   = (float*)alloc((size_t)B_ * EC_ * T1s * 4);
    float*  cmean  = (float*)alloc(256 * 4);
    float*  crstd  = (float*)alloc(256 * 4);
    const size_t NRs = (size_t)B_ * NE_ * TRs;
    float*  yo   = (float*)alloc(NRs * 4);
    float*  h0   = (float*)alloc(NRs * 4);
    __bf16* hnbf = (__bf16*)alloc(NRs * 2);
    float*  kkb  = (float*)alloc(NRs * 4);
    float*  vvb  = (float*)alloc(NRs * 4);
    float*  srb  = (float*)alloc(NRs * 4);
    __bf16* wsrb = (__bf16*)alloc(NRs * 2);
    float*  h1   = (float*)alloc(NRs * 4);
    __bf16* hn2b = (__bf16*)alloc(NRs * 2);
    __bf16* kfb  = (__bf16*)alloc((size_t)B_ * HID_ * TRs * 2);
    float*  ffnv = (float*)alloc(NRs * 4);
    float*  rr2  = (float*)alloc(NRs * 4);

    auto gemm = [&](const float* Wt, int KT, int WT, const __bf16* X, const float* bias,
                    const float* addin, float* Yf, __bf16* Ybf, int M, int T, int Ts, int act) {
        dim3 g(Ts / 64, M / 16, B_);
        if (!WT && KT == 64)
            k_wmma_gemm<64, false><<<g, 128, 0, stream>>>(Wt, X, bias, addin, Yf, Ybf, M, T, Ts, act);
        else if (!WT && KT == 128)
            k_wmma_gemm<128, false><<<g, 128, 0, stream>>>(Wt, X, bias, addin, Yf, Ybf, M, T, Ts, act);
        else if (WT && KT == 64)
            k_wmma_gemm<64, true><<<g, 128, 0, stream>>>(Wt, X, bias, addin, Yf, Ybf, M, T, Ts, act);
        else /* WT && KT == 256 */
            k_wmma_gemm<256, true><<<g, 128, 0, stream>>>(Wt, X, bias, addin, Yf, Ybf, M, T, Ts, act);
    };

    // 1) BN over time index
    k_tstats<<<(T0_ + 255) / 256, 256, 0, stream>>>(spike, mean_t, rstd_t);
    {
        size_t N = (size_t)B_ * E_ * T0_;
        k_bn1_apply<<<(int)((N + 255) / 256), 256, 0, stream>>>(spike, mean_t, rstd_t, bn1g, bn1b, xbf);
    }
    // 2) Chebyshev combined matrix, then one big WMMA GEMM with fused ReLU
    k_cheb<<<1, 256, 0, stream>>>(Amat, gcw, gcb, Lm, T2m, Mm, gbias);
    gemm(Mm, 128, 0, xbf, gbias, nullptr, nullptr, goutbf, E_, T0_, T0_, /*relu*/1);
    // 3) projection conv (im2col WMMA GEMM)
    {
        dim3 g(TPs_ / 64, FC_ / 16, B_);
        k_wmma_proj<<<g, 128, 0, stream>>>(projw, goutbf, y1);
    }
    // 4) cLN -> bf16
    k_lnch<<<(B_ * TP_ + 255) / 256, 256, 0, stream>>>(y1, clng, clnb, nullptr, ybf, FC_, TP_, TPs_, B_);
    // 5) 1x1 conv in
    gemm(inw, 64, 0, ybf, inb, nullptr, yA, yAbf, FC_, TP_, TPs_, 0);

    // 6) ResBlocks
    struct RB { const float *c1,*b1g,*b1b,*c2,*b2g,*b2b,*a1,*a2,*dn; int Cin,Cout,T,Ts,To,Tos; };
    RB rbs[3] = {
        { r0c1,r0b1g,r0b1b,r0c2,r0b2g,r0b2b,r0a1,r0a2,nullptr, FC_,FC_,TP_,TPs_,T1,T1s },
        { r1c1,r1b1g,r1b1b,r1c2,r1b2g,r1b2b,r1a1,r1a2,r1dn,    FC_,EC_,T1,T1s,T2,T2s },
        { r2c1,r2b1g,r2b1b,r2c2,r2b2g,r2b2b,r2a1,r2a2,nullptr, EC_,EC_,T2,T2s,TR,TRs },
    };
    float* curF = yA; __bf16* curB = yAbf;
    float* nxtF = yB; __bf16* nxtB = yBbf;
    for (int r = 0; r < 3; ++r) {
        RB& p = rbs[r];
        size_t N1 = (size_t)B_ * p.Cout * p.T;
        gemm(p.c1, p.Cin, 0, curB, nullptr, nullptr, t1, nullptr, p.Cout, p.T, p.Ts, 0);
        k_chanstats<<<p.Cout, 256, 0, stream>>>(t1, cmean, crstd, p.Cout, p.T, p.Ts, B_);
        k_bnprelu<<<(int)((N1 + 255) / 256), 256, 0, stream>>>(t1, cmean, crstd, p.b1g, p.b1b, p.a1, zbf, p.Cout, p.T, p.Ts, N1);
        gemm(p.c2, p.Cout, 0, zbf, nullptr, nullptr, t2, nullptr, p.Cout, p.T, p.Ts, 0);
        k_chanstats<<<p.Cout, 256, 0, stream>>>(t2, cmean, crstd, p.Cout, p.T, p.Ts, B_);
        const float* resid = curF;
        if (p.dn) {
            gemm(p.dn, p.Cin, 0, curB, nullptr, nullptr, dres, nullptr, p.Cout, p.T, p.Ts, 0);
            resid = dres;
        }
        size_t No = (size_t)B_ * p.Cout * p.To;
        k_resmerge<<<(int)((No + 255) / 256), 256, 0, stream>>>(t2, cmean, crstd, p.b2g, p.b2b,
                                                               resid, p.a2, nxtF, nxtB,
                                                               p.Cout, p.Ts, p.To, p.Tos, B_);
        float* tf = curF; curF = nxtF; nxtF = tf;
        __bf16* tb = curB; curB = nxtB; nxtB = tb;
    }

    // 7) 1x1 conv out (128 -> 64)
    gemm(outw, EC_, 0, curB, outb, nullptr, yo, nullptr, NE_, TR, TRs, 0);

    // 8) RWKV block (channel-major [B,C,TRs], C=64, T=270)
    k_lnch<<<(B_ * TR + 255) / 256, 256, 0, stream>>>(yo, ln0g, ln0b, h0, nullptr, NE_, TR, TRs, B_);
    k_lnch<<<(B_ * TR + 255) / 256, 256, 0, stream>>>(h0, ln1g, ln1b, nullptr, hnbf, NE_, TR, TRs, B_);
    gemm(Wk, NE_, 1, hnbf, nullptr, nullptr, kkb, nullptr, NE_, TR, TRs, 0);
    gemm(Wv, NE_, 1, hnbf, nullptr, nullptr, vvb, nullptr, NE_, TR, TRs, 0);
    gemm(Wr, NE_, 1, hnbf, nullptr, nullptr, srb, nullptr, NE_, TR, TRs, 2);
    k_wkv<<<B_ * NE_, 128, 0, stream>>>(kkb, vvb, srb, decay, first, wsrb, NE_, TR, TRs);
    gemm(Wo, NE_, 1, wsrb, nullptr, h0, h1, nullptr, NE_, TR, TRs, 0);
    // channel mix (FFN)
    k_lnch<<<(B_ * TR + 255) / 256, 256, 0, stream>>>(h1, ln2g, ln2b, nullptr, hn2b, NE_, TR, TRs, B_);
    gemm(fWk, NE_, 1, hn2b, nullptr, nullptr, nullptr, kfb, HID_, TR, TRs, 3);
    gemm(fWv, 256, 1, kfb, nullptr, nullptr, ffnv, nullptr, NE_, TR, TRs, 0);
    gemm(fWr, NE_, 1, hn2b, nullptr, nullptr, rr2, nullptr, NE_, TR, TRs, 2);
    k_final<<<(NRlog + 255) / 256, 256, 0, stream>>>(h1, rr2, ffnv, (float*)d_out, TR, TRs, NRlog);
}
